// NodeEdge_35201551958055
// MI455X (gfx1250) — matrix operand order repaired and checked by reading, not compiled, not run on hardware
//
#include <hip/hip_runtime.h>

typedef _Float16 v8h  __attribute__((ext_vector_type(8)));
typedef _Float16 v16h __attribute__((ext_vector_type(16)));
typedef float    v4f  __attribute__((ext_vector_type(4)));
typedef float    v8f  __attribute__((ext_vector_type(8)));

#define B_   16
#define N_   2048
#define E_   8192
#define DIM_ 64
#define DH_  32
#define C_   (B_ * DH_)      // 512 fused (b,h) columns

#define KSTEP   64                // K depth per LDS panel (two 32-wide WMMA steps)
#define NSUP    (E_ / KSTEP)      // 128 super-steps
#define LSTRIDE 72                // 64 halfs + 8 pad = 144 B row (9 x 16B granules)

// ---------------------------------------------------------------------------
// Phase 1: xeH[c = b*32+h][e] = f16( relu( sum_d inputs[b,e,d]*W[d,h] + bias[h] ) )
// K(=e)-contiguous rows feed the WMMA B-operand directly. Lanes of a wave share
// one input row (broadcast loads); W column reads are lane-coalesced.
// ---------------------------------------------------------------------------
__global__ __launch_bounds__(256) void xe_kernel(
    const float* __restrict__ in,    // [B,E,DIM] f32
    const float* __restrict__ W,     // [DIM,DH] f32
    const float* __restrict__ bias,  // [DH] f32
    _Float16* __restrict__ xeH)      // [C_, E_] f16
{
    int tid  = blockIdx.x * blockDim.x + threadIdx.x;
    int h    = tid & (DH_ - 1);
    int row  = tid >> 5;            // b*E + e
    int b    = row >> 13;
    int e    = row & (E_ - 1);
    const float* r = in + (size_t)row * DIM_;
    float acc = bias[h];
#pragma unroll
    for (int d = 0; d < DIM_; ++d)
        acc = fmaf(r[d], W[d * DH_ + h], acc);
    acc = acc > 0.0f ? acc : 0.0f;
    xeH[(size_t)(b * DH_ + h) * E_ + e] = (_Float16)acc;
}

// ---------------------------------------------------------------------------
// Phase 2: Ah[n][e] = f16( w*inci + b ). Streams 192 MB of f32 exactly once
// (the HBM-bound part of the whole problem); b128 loads, b128 f16 store.
// ---------------------------------------------------------------------------
__global__ __launch_bounds__(256) void a_kernel(
    const float* __restrict__ w, const float* __restrict__ inci,
    const float* __restrict__ bb, _Float16* __restrict__ Ah)
{
    size_t i = ((size_t)blockIdx.x * blockDim.x + threadIdx.x) * 8;
    v4f w0 = *(const v4f*)(w + i),    w1 = *(const v4f*)(w + i + 4);
    v4f m0 = *(const v4f*)(inci + i), m1 = *(const v4f*)(inci + i + 4);
    v4f b0 = *(const v4f*)(bb + i),   b1 = *(const v4f*)(bb + i + 4);
    v8h o;
#pragma unroll
    for (int j = 0; j < 4; ++j) {
        o[j]     = (_Float16)fmaf(w0[j], m0[j], b0[j]);
        o[j + 4] = (_Float16)fmaf(w1[j], m1[j], b1[j]);
    }
    *(v8h*)(Ah + i) = o;
}

// ---------------------------------------------------------------------------
// Phase 3: Out2[2048,512] = Ah[2048,8192] x XeH^T via v_wmma_f32_16x16x32_f16.
// Block (8 waves) owns a 64x64 output tile. Per super-step, 64x64 f16 A and B
// panels are double-buffered in LDS; source order is
//     issue global loads -> compute (ds_load frags + 4 WMMAs) -> ds_store -> barrier
// so s_wait_loadcnt falls AFTER the WMMAs (latency overlapped). Manual 2x
// unroll keeps buffer parity literal (no cndmask address fixups).
// Fragment layouts per CDNA5 ISA:
//   A 16x32 f16 : lane<16 row=lane, K {0..7}+{16..23}; lane>=16 K {8..15}+{24..31}
//   B 32x16 f16 : lane<16 col=lane, K {0..7}+{16..23}; lane>=16 K {8..15}+{24..31}
//   C/D f32     : c[r] = (M = half*8 + r, N = lane%16)
// ---------------------------------------------------------------------------
__global__ __launch_bounds__(256) void gemm_kernel(
    const _Float16* __restrict__ Ah,   // [N_, E_]
    const _Float16* __restrict__ XeH,  // [C_, E_]
    float* __restrict__ out)           // [B_, N_, DH_]
{
    __shared__ __align__(16) _Float16 As[2][64 * LSTRIDE];
    __shared__ __align__(16) _Float16 Bs[2][64 * LSTRIDE];

    const int tid  = threadIdx.x;
    const int lane = tid & 31;
    const int wave = tid >> 5;
    const int lrow = lane & 15;
    const int half = lane >> 4;

    const int m_blk = (blockIdx.x >> 3) * 64;   // 32 m-blocks
    const int n_blk = (blockIdx.x & 7) * 64;    // 8  n-blocks

    const int m_loc = (wave & 3) * 16;
    const int n_loc = (wave >> 2) * 32;

    // --- staging map: each thread owns a contiguous 64B run of one panel row ---
    const int sArr = tid >> 7;                  // 0: A panel, 1: B panel (wave-uniform)
    const int sRow = (tid & 127) >> 1;          // 0..63
    const int sCol = (tid & 1) * 32;            // half offset (64 B)
    const _Float16* sgp = (sArr ? XeH + (size_t)(n_blk + sRow) * E_
                                : Ah  + (size_t)(m_blk + sRow) * E_) + sCol;
    _Float16* const slds[2] = {
        (sArr ? Bs[0] : As[0]) + sRow * LSTRIDE + sCol,
        (sArr ? Bs[1] : As[1]) + sRow * LSTRIDE + sCol };

    struct Regs { v8h x0, x1, x2, x3; };
    auto stage_load = [&](int k0) -> Regs {
        Regs r;
        r.x0 = *(const v8h*)(sgp + k0);
        r.x1 = *(const v8h*)(sgp + k0 + 8);
        r.x2 = *(const v8h*)(sgp + k0 + 16);
        r.x3 = *(const v8h*)(sgp + k0 + 24);
        return r;
    };
    auto stage_store = [&](const Regs& r, _Float16* d) {
        *(v8h*)(d)      = r.x0;
        *(v8h*)(d + 8)  = r.x1;
        *(v8h*)(d + 16) = r.x2;
        *(v8h*)(d + 24) = r.x3;
    };

    v8f c0 = {}; v8f c1 = {};
    auto compute = [&](const _Float16* Ab, const _Float16* Bb) {
#pragma unroll
        for (int s = 0; s < 2; ++s) {           // two 32-wide K sub-steps
            union { v16h v; v8h h[2]; } a, bA, bB;
            const _Float16* pa = Ab + (m_loc + lrow) * LSTRIDE + s * 32 + half * 8;
            a.h[0] = *(const v8h*)(pa);
            a.h[1] = *(const v8h*)(pa + 16);
            const _Float16* pb = Bb + (n_loc + lrow) * LSTRIDE + s * 32 + half * 8;
            bA.h[0] = *(const v8h*)(pb);
            bA.h[1] = *(const v8h*)(pb + 16);
            const _Float16* pq = pb + 16 * LSTRIDE;
            bB.h[0] = *(const v8h*)(pq);
            bB.h[1] = *(const v8h*)(pq + 16);
            c0 = __builtin_amdgcn_wmma_f32_16x16x32_f16(
                    false, a.v, false, bA.v, (short)0, c0, false, false);
            c1 = __builtin_amdgcn_wmma_f32_16x16x32_f16(
                    false, a.v, false, bB.v, (short)0, c1, false, false);
        }
    };

    // prologue: fill buffer 0 with super-step 0
    {
        Regs r0 = stage_load(0);
        stage_store(r0, slds[0]);
    }
    __syncthreads();

    for (int kk = 0; kk < NSUP; kk += 2) {
        // ---- even step: compute buf0, stage kk+1 into buf1 ----
        Regs rA = stage_load((kk + 1) * KSTEP);      // kk+1 <= 127: always valid
        compute(As[0], Bs[0]);
        stage_store(rA, slds[1]);
        __syncthreads();
        // ---- odd step: compute buf1, stage kk+2 into buf0 ----
        int k2 = (kk + 2 < NSUP) ? (kk + 2) * KSTEP : 0;  // dummy restage on last
        Regs rB = stage_load(k2);
        compute(As[1], Bs[1]);
        stage_store(rB, slds[0]);
        __syncthreads();
    }

    // --- scatter D tiles to out[b][node][h], col = b*32 + h ---
    int col0 = n_blk + n_loc + lrow;
    int col1 = col0 + 16;
    int b0i = col0 >> 5, h0 = col0 & 31;
    int b1i = col1 >> 5, h1 = col1 & 31;
#pragma unroll
    for (int r = 0; r < 8; ++r) {
        int m = m_blk + m_loc + half * 8 + r;   // node index
        out[((size_t)b0i * N_ + m) * DH_ + h0] = c0[r];
        out[((size_t)b1i * N_ + m) * DH_ + h1] = c1[r];
    }
}

extern "C" void kernel_launch(void* const* d_in, const int* in_sizes, int n_in,
                              void* d_out, int out_size, void* d_ws, size_t ws_size,
                              hipStream_t stream) {
    const float* inputs = (const float*)d_in[0];  // [B,E,DIM]
    const float* W      = (const float*)d_in[1];  // [DIM,DH]
    const float* bx     = (const float*)d_in[2];  // [DH]
    const float* inci   = (const float*)d_in[3];  // [N,E]
    const float* w      = (const float*)d_in[4];  // [N,E]
    const float* bb     = (const float*)d_in[5];  // [N,E]
    float* out          = (float*)d_out;

    _Float16* xeH = (_Float16*)d_ws;                                   // 8 MB
    _Float16* Ah  = (_Float16*)((char*)d_ws + (size_t)C_ * E_ * 2);    // 32 MB

    xe_kernel<<<(B_ * E_ * DH_) / 256, 256, 0, stream>>>(inputs, W, bx, xeH);
    a_kernel<<<((size_t)N_ * E_ / 8) / 256, 256, 0, stream>>>(w, inci, bb, Ah);
    gemm_kernel<<<(N_ / 64) * (C_ / 64), 256, 0, stream>>>(Ah, xeH, out);
}
